// ModelXy_4088808866451
// MI455X (gfx1250) — compile-verified, matrix operand order checked by reading, not run on hardware
//
#include <hip/hip_runtime.h>

// ---------------------------------------------------------------------------
// Types matching gfx1250 WMMA operand shapes
// ---------------------------------------------------------------------------
typedef __attribute__((ext_vector_type(16))) __bf16        v16bf;
typedef __attribute__((ext_vector_type(8)))  float         v8f;
typedef __attribute__((ext_vector_type(4)))  unsigned int  uv4;
typedef __attribute__((ext_vector_type(2)))  unsigned int  uv2;
typedef __attribute__((ext_vector_type(4)))  float         fv4;

#define S_LEN  512
#define BATCH  64
#define EMBD   256
#define HID    512
#define NGATE  2048              // 4*HID
#define NTAG   5
#define MROWS  (S_LEN * BATCH)   // 32768
#define LDSROW 520               // ushorts per h row: 512 + 8 pad (1040B = 65*16B)

union FragBF { v16bf v; uv4 u[2]; };

__device__ __forceinline__ unsigned short f2bf(float f) {
  unsigned int u = __float_as_uint(f);
  u += 0x7FFFu + ((u >> 16) & 1u);          // round-to-nearest-even
  return (unsigned short)(u >> 16);
}
__device__ __forceinline__ float sigmoidf_(float x) {
  return 1.0f / (1.0f + __expf(-x));
}
__device__ __forceinline__ v8f v8f_zero() {
  v8f z = {0.f,0.f,0.f,0.f,0.f,0.f,0.f,0.f};
  return z;
}

// ---------------------------------------------------------------------------
// One-time prep: bias sums, padded projection weights, loss accumulators
// ---------------------------------------------------------------------------
__global__ void k_init(const float* __restrict__ bih_f, const float* __restrict__ bhh_f,
                       const float* __restrict__ bih_b, const float* __restrict__ bhh_b,
                       const float* __restrict__ sbj_w, const float* __restrict__ sbj_b,
                       const float* __restrict__ obj_w, const float* __restrict__ obj_b,
                       float* __restrict__ bsumF, float* __restrict__ bsumB,
                       unsigned short* __restrict__ Wproj, float* __restrict__ bproj,
                       float* __restrict__ wsloss) {
  const int tid = blockIdx.x * 256 + threadIdx.x;
  if (tid < NGATE) {
    bsumF[tid] = bih_f[tid] + bhh_f[tid];
    bsumB[tid] = bih_b[tid] + bhh_b[tid];
  }
  if (tid < 16 * 2 * HID) {                 // padded [16, 1024] projection, rows 0-4 sbj, 5-9 obj
    const int r = tid >> 10, k = tid & 1023;
    float v = (r < NTAG) ? sbj_w[r * 2 * HID + k]
            : (r < 2 * NTAG) ? obj_w[(r - NTAG) * 2 * HID + k] : 0.0f;
    Wproj[tid] = f2bf(v);
  }
  if (tid < 16) bproj[tid] = (tid < NTAG) ? sbj_b[tid]
                           : (tid < 2 * NTAG) ? obj_b[tid - NTAG] : 0.0f;
  if (tid < 8) wsloss[tid] = 0.0f;
}

__global__ void k_cast(const float* __restrict__ src, unsigned short* __restrict__ dst, int n) {
  const int i = blockIdx.x * 256 + threadIdx.x;
  if (i < n) dst[i] = f2bf(src[i]);
}

// ---------------------------------------------------------------------------
// Embedding gather + cast: Xbf[s*64+b][0:256] = bf16(emb[text[b][s]])
// ---------------------------------------------------------------------------
__global__ void k_embed(const int* __restrict__ text, const float* __restrict__ emb,
                        unsigned short* __restrict__ Xbf) {
  const size_t tid = (size_t)blockIdx.x * 256 + threadIdx.x;   // exactly MROWS*64 threads
  const int    e4  = (int)(tid & 63);
  const size_t r   = tid >> 6;           // r = s*64 + b
  const int    s   = (int)(r >> 6);
  const int    b   = (int)(r & 63);
  const int    idx = text[b * S_LEN + s];
  const fv4 v = *(const fv4*)(emb + (size_t)idx * EMBD + e4 * 4);
  uv2 pk;
  pk.x = (unsigned int)f2bf(v.x) | ((unsigned int)f2bf(v.y) << 16);
  pk.y = (unsigned int)f2bf(v.z) | ((unsigned int)f2bf(v.w) << 16);
  *(uv2*)(Xbf + r * EMBD + e4 * 4) = pk;
}

// ---------------------------------------------------------------------------
// Persistent BiLSTM: 2 blocks (fwd/bwd) x 1024 threads (32 waves).
// Per step: gates[64,2048] = [X_t | h]@[Wih|Whh]^T via WMMA bf16; cell update
// in registers; h ping-pongs in LDS; c lives in VGPRs; h-history to global.
//
// Addressing: exactly FOUR base pointers per step (X row, Wih row, Whh row,
// LDS h row) with the lane-dependent part folded in; every gate / tile-group
// / K-chunk variation is a compile-time byte offset that fits the 24-bit
// global IOFFSET (gate stride <= 1.5MB) or 16-bit DS offset (<= 50KB), so no
// per-gate 64-bit address registers exist to spill. The opaque wofs (always
// 0) is renewed per step so LICM cannot hoist the weight loads out of the
// 512-step loop, while address-space inference still yields global_load_b128.
// ---------------------------------------------------------------------------
__global__ __launch_bounds__(1024) void k_lstm(
    const unsigned short* __restrict__ Xbf,
    const unsigned short* __restrict__ WihF, const unsigned short* __restrict__ WhhF,
    const unsigned short* __restrict__ WihB, const unsigned short* __restrict__ WhhB,
    const float* __restrict__ bsumF, const float* __restrict__ bsumB,
    unsigned short* __restrict__ vec)                     // [S][B][2H] bf16
{
  extern __shared__ unsigned short hbuf[];                // 2 x [64][LDSROW]
  const int dir = blockIdx.x;
  const unsigned short* Wih  = dir ? WihB  : WihF;
  const unsigned short* Whh  = dir ? WhhB  : WhhF;
  const float*          bsum = dir ? bsumB : bsumF;
  const int tid  = threadIdx.x;
  const int wave = tid >> 5;
  const int lane = tid & 31;
  const int lr   = lane & 15;      // tile row (A) / col (B,D)
  const int lh   = lane >> 4;      // K-half select per ISA fragment layout
  const int lofs = lh * 8;         // lane's K-phase (elements)
  const int col  = wave * 16 + lr; // this lane's hidden column (same for all tgs)

  for (int i = tid; i < 2 * BATCH * LDSROW; i += 1024) hbuf[i] = 0;
  float creg[4][8];
  #pragma unroll
  for (int tg = 0; tg < 4; ++tg)
    #pragma unroll
    for (int r = 0; r < 8; ++r) creg[tg][r] = 0.0f;

  // Loop-invariant gate biases for this lane's column (4 VGPRs).
  const float bi  = bsum[0 * HID + col];
  const float bf_ = bsum[1 * HID + col];
  const float bg  = bsum[2 * HID + col];
  const float bo  = bsum[3 * HID + col];
  __syncthreads();

  for (int step = 0; step < S_LEN; ++step) {
    const int s = dir ? (S_LEN - 1 - step) : step;
    const unsigned short* hrd = hbuf + (size_t)(step & 1) * (BATCH * LDSROW);
    unsigned short*       hwr = hbuf + (size_t)((step + 1) & 1) * (BATCH * LDSROW);

    // Opaque zero offset, renewed every iteration: defeats LICM on the weight
    // fragment loads without breaking global address-space inference.
    unsigned int wofs = 0;
    asm("" : "+s"(wofs));

    // The only four load base pointers of the inner loops.
    const unsigned short* arowX = Xbf + ((size_t)(s * BATCH) + lr) * EMBD + lofs;
    const unsigned short* browI = Wih + wofs + (size_t)col * EMBD + lofs;
    const unsigned short* browH = Whh + wofs + (size_t)col * HID  + lofs;
    const unsigned short* arowH = hrd + (size_t)lr * LDSROW + lofs;
    // Store bases (constant-offset stores off these).
    unsigned short* hwrL = hwr + (size_t)(lh * 8) * LDSROW + col;
    unsigned short* vrow = vec + ((size_t)s * BATCH + lh * 8) * (2 * HID)
                               + (size_t)dir * HID + col;

    if (step + 1 < S_LEN) {      // pull next step's activations toward the WGP
      const int sn = dir ? (S_LEN - 2 - step) : (step + 1);
      __builtin_prefetch(Xbf + ((size_t)(sn * BATCH) + (tid & 63)) * EMBD, 0, 1);
    }

    #pragma unroll   // keep creg[tg][*] register-resident
    for (int tg = 0; tg < 4; ++tg) {
      v8f acc[4];
      #pragma unroll
      for (int g = 0; g < 4; ++g) acc[g] = v8f_zero();

      // ---- input projection part: K = EMBD, A from global (row-major) ----
      #pragma unroll
      for (int kc = 0; kc < EMBD / 32; ++kc) {
        FragBF a;
        a.u[0] = *(const uv4*)(arowX + tg * 16 * EMBD + kc * 32);
        a.u[1] = *(const uv4*)(arowX + tg * 16 * EMBD + kc * 32 + 16);
        #pragma unroll
        for (int g = 0; g < 4; ++g) {
          FragBF b;
          b.u[0] = *(const uv4*)(browI + g * HID * EMBD + kc * 32);
          b.u[1] = *(const uv4*)(browI + g * HID * EMBD + kc * 32 + 16);
          acc[g] = __builtin_amdgcn_wmma_f32_16x16x32_bf16(
                     false, a.v, false, b.v, (short)0, acc[g], false, false);
        }
      }
      // ---- recurrent part: K = HID, A from LDS (h_{t-1}) ----
      #pragma unroll
      for (int kc = 0; kc < HID / 32; ++kc) {
        FragBF a;
        a.u[0] = *(const uv4*)(arowH + tg * 16 * LDSROW + kc * 32);
        a.u[1] = *(const uv4*)(arowH + tg * 16 * LDSROW + kc * 32 + 16);
        #pragma unroll
        for (int g = 0; g < 4; ++g) {
          FragBF b;
          b.u[0] = *(const uv4*)(browH + g * HID * HID + kc * 32);
          b.u[1] = *(const uv4*)(browH + g * HID * HID + kc * 32 + 16);
          acc[g] = __builtin_amdgcn_wmma_f32_16x16x32_bf16(
                     false, a.v, false, b.v, (short)0, acc[g], false, false);
        }
      }
      // ---- LSTM cell, entirely in this lane's registers ----
      #pragma unroll
      for (int r = 0; r < 8; ++r) {
        const float iv = sigmoidf_(acc[0][r] + bi);
        const float fv = sigmoidf_(acc[1][r] + bf_);
        const float gv = tanhf   (acc[2][r] + bg);
        const float ov = sigmoidf_(acc[3][r] + bo);
        const float c  = fv * creg[tg][r] + iv * gv;
        creg[tg][r] = c;
        const float h  = ov * tanhf(c);
        const unsigned short hb = f2bf(h);
        hwrL[(size_t)(tg * 16 + r) * LDSROW] = hb;        // constant DS offsets
        vrow[(size_t)(tg * 16 + r) * (2 * HID)] = hb;     // constant IOFFSETs
      }
    }
    __syncthreads();   // h_t fully in LDS before any wave starts step t+1
  }
}

// ---------------------------------------------------------------------------
// Emissions: em[32768,16] = vec[32768,1024] @ Wproj^T + bproj (N padded to 16)
// ---------------------------------------------------------------------------
__global__ __launch_bounds__(256) void k_proj(const unsigned short* __restrict__ vec,
                                              const unsigned short* __restrict__ Wproj,
                                              const float* __restrict__ bproj,
                                              float* __restrict__ em) {
  const int w    = (blockIdx.x * 256 + threadIdx.x) >> 5;   // 2048 waves, one 16x16 tile each
  const int lane = threadIdx.x & 31;
  const int lr   = lane & 15;
  const int lh   = lane >> 4;
  const int m0   = w * 16;
  v8f acc = v8f_zero();
  const unsigned short* arow = vec   + (size_t)(m0 + lr) * (2 * HID) + lh * 8;
  const unsigned short* brow = Wproj + (size_t)lr * (2 * HID) + lh * 8;
  #pragma unroll
  for (int kc = 0; kc < (2 * HID) / 32; ++kc) {
    FragBF a, b;
    a.u[0] = *(const uv4*)(arow + kc * 32);
    a.u[1] = *(const uv4*)(arow + kc * 32 + 16);
    b.u[0] = *(const uv4*)(brow + kc * 32);
    b.u[1] = *(const uv4*)(brow + kc * 32 + 16);
    acc = __builtin_amdgcn_wmma_f32_16x16x32_bf16(
            false, a.v, false, b.v, (short)0, acc, false, false);
  }
  const float bias = bproj[lr];
  #pragma unroll
  for (int r = 0; r < 8; ++r)
    em[(size_t)(m0 + r + lh * 8) * 16 + lr] = acc[r] + bias;
}

// ---------------------------------------------------------------------------
// CRF token-mean NLL: block 0 = sbj, block 1 = obj; one thread per batch row.
// ---------------------------------------------------------------------------
__global__ void k_crf(const float* __restrict__ em, const int* __restrict__ text,
                      const int* __restrict__ sbjs, const int* __restrict__ objs,
                      const float* __restrict__ sbj_start, const float* __restrict__ sbj_end,
                      const float* __restrict__ sbj_trans,
                      const float* __restrict__ obj_start, const float* __restrict__ obj_end,
                      const float* __restrict__ obj_trans,
                      float* __restrict__ wsloss) {
  const int which = blockIdx.x;
  const int b     = threadIdx.x;
  const float* startp = which ? obj_start : sbj_start;
  const float* endp   = which ? obj_end   : sbj_end;
  const float* transp = which ? obj_trans : sbj_trans;
  const int*   tags   = which ? objs      : sbjs;
  const int    base   = which * NTAG;

  __shared__ float tr[NTAG * NTAG];
  __shared__ float rednll[BATCH];
  __shared__ float redtok[BATCH];
  if (b < NTAG * NTAG) tr[b] = transp[b];
  __syncthreads();

  float st[NTAG], en[NTAG];
  #pragma unroll
  for (int j = 0; j < NTAG; ++j) { st[j] = startp[j]; en[j] = endp[j]; }

  const int t0 = tags[b * S_LEN + 0];
  float alpha[NTAG];
  {
    const float* e0 = em + (size_t)b * 16 + base;
    #pragma unroll
    for (int j = 0; j < NTAG; ++j) alpha[j] = st[j] + e0[j];
  }
  float score  = alpha[t0];                       // start[t0] + em[0,b,t0]
  int   prev   = t0;
  int   tokens = (text[b * S_LEN + 0] != 0) ? 1 : 0;

  for (int s = 1; s < S_LEN; ++s) {
    const int m = (text[b * S_LEN + s] != 0) ? 1 : 0;
    const int t = tags[b * S_LEN + s];
    const float* e = em + ((size_t)s * BATCH + b) * 16 + base;
    if (m) {
      score += tr[prev * NTAG + t] + e[t];
      ++tokens;
      float na[NTAG];
      #pragma unroll
      for (int j = 0; j < NTAG; ++j) {
        float mx = alpha[0] + tr[0 * NTAG + j];
        #pragma unroll
        for (int i = 1; i < NTAG; ++i) mx = fmaxf(mx, alpha[i] + tr[i * NTAG + j]);
        float ss = 0.0f;
        #pragma unroll
        for (int i = 0; i < NTAG; ++i) ss += __expf(alpha[i] + tr[i * NTAG + j] - mx);
        na[j] = mx + __logf(ss) + e[j];
      }
      #pragma unroll
      for (int j = 0; j < NTAG; ++j) alpha[j] = na[j];
    }
    prev = t;
  }
  score += en[tags[b * S_LEN + (tokens - 1)]];    // end[tags[seq_ends]]

  float mx = alpha[0] + en[0];
  #pragma unroll
  for (int j = 1; j < NTAG; ++j) mx = fmaxf(mx, alpha[j] + en[j]);
  float ss = 0.0f;
  #pragma unroll
  for (int j = 0; j < NTAG; ++j) ss += __expf(alpha[j] + en[j] - mx);
  const float logZ = mx + __logf(ss);

  rednll[b] = -(score - logZ);
  redtok[b] = (float)tokens;
  __syncthreads();
  if (b == 0) {
    float sn = 0.0f, stk = 0.0f;
    for (int i = 0; i < BATCH; ++i) { sn += rednll[i]; stk += redtok[i]; }
    wsloss[which] = sn / stk;                     // token_mean NLL
  }
}

__global__ void k_final(const float* __restrict__ wsloss, float* __restrict__ out) {
  if (threadIdx.x == 0 && blockIdx.x == 0) {
    const float a = wsloss[0], b = wsloss[1];
    out[0] = 2.0f * (a + b);
    out[1] = a;
    out[2] = b;
  }
}

// ---------------------------------------------------------------------------
extern "C" void kernel_launch(void* const* d_in, const int* in_sizes, int n_in,
                              void* d_out, int out_size, void* d_ws, size_t ws_size,
                              hipStream_t stream) {
  (void)in_sizes; (void)n_in; (void)out_size; (void)ws_size;
  const int*   text      = (const int*)  d_in[0];
  const int*   sbjs      = (const int*)  d_in[1];
  const int*   objs      = (const int*)  d_in[2];
  const float* emb       = (const float*)d_in[3];
  const float* wih_f     = (const float*)d_in[4];
  const float* whh_f     = (const float*)d_in[5];
  const float* bih_f     = (const float*)d_in[6];
  const float* bhh_f     = (const float*)d_in[7];
  const float* wih_b     = (const float*)d_in[8];
  const float* whh_b     = (const float*)d_in[9];
  const float* bih_b     = (const float*)d_in[10];
  const float* bhh_b     = (const float*)d_in[11];
  const float* sbj_w     = (const float*)d_in[12];
  const float* sbj_b     = (const float*)d_in[13];
  const float* obj_w     = (const float*)d_in[14];
  const float* obj_b     = (const float*)d_in[15];
  const float* sbj_start = (const float*)d_in[16];
  const float* sbj_end   = (const float*)d_in[17];
  const float* sbj_trans = (const float*)d_in[18];
  const float* obj_start = (const float*)d_in[19];
  const float* obj_end   = (const float*)d_in[20];
  const float* obj_trans = (const float*)d_in[21];

  char* ws = (char*)d_ws;
  size_t off = 0;
  auto alloc = [&](size_t bytes) -> void* {
    void* p = ws + off;
    off = (off + bytes + 255) & ~(size_t)255;
    return p;
  };
  unsigned short* Xbf   = (unsigned short*)alloc((size_t)MROWS * EMBD * 2);        // 16 MB
  unsigned short* WihFb = (unsigned short*)alloc((size_t)NGATE * EMBD * 2);
  unsigned short* WhhFb = (unsigned short*)alloc((size_t)NGATE * HID  * 2);
  unsigned short* WihBb = (unsigned short*)alloc((size_t)NGATE * EMBD * 2);
  unsigned short* WhhBb = (unsigned short*)alloc((size_t)NGATE * HID  * 2);
  float*          bsumF = (float*)         alloc((size_t)NGATE * 4);
  float*          bsumB = (float*)         alloc((size_t)NGATE * 4);
  unsigned short* Wproj = (unsigned short*)alloc((size_t)16 * 2 * HID * 2);
  float*          bproj = (float*)         alloc(16 * 4);
  unsigned short* vecb  = (unsigned short*)alloc((size_t)MROWS * 2 * HID * 2);     // 64 MB
  float*          emf   = (float*)         alloc((size_t)MROWS * 16 * 4);          // 2 MB
  float*          wsl   = (float*)         alloc(8 * 4);

  k_init<<<64, 256, 0, stream>>>(bih_f, bhh_f, bih_b, bhh_b,
                                 sbj_w, sbj_b, obj_w, obj_b,
                                 bsumF, bsumB, Wproj, bproj, wsl);
  k_cast<<<(NGATE * EMBD + 255) / 256, 256, 0, stream>>>(wih_f, WihFb, NGATE * EMBD);
  k_cast<<<(NGATE * HID  + 255) / 256, 256, 0, stream>>>(whh_f, WhhFb, NGATE * HID);
  k_cast<<<(NGATE * EMBD + 255) / 256, 256, 0, stream>>>(wih_b, WihBb, NGATE * EMBD);
  k_cast<<<(NGATE * HID  + 255) / 256, 256, 0, stream>>>(whh_b, WhhBb, NGATE * HID);
  k_embed<<<(MROWS * 64) / 256, 256, 0, stream>>>(text, emb, Xbf);

  const size_t lstm_lds = (size_t)2 * BATCH * LDSROW * sizeof(unsigned short);     // ~130 KB of 320 KB/WGP
  k_lstm<<<2, 1024, lstm_lds, stream>>>(Xbf, WihFb, WhhFb, WihBb, WhhBb, bsumF, bsumB, vecb);

  k_proj<<<256, 256, 0, stream>>>(vecb, Wproj, bproj, emf);
  k_crf<<<2, BATCH, 0, stream>>>(emf, text, sbjs, objs,
                                 sbj_start, sbj_end, sbj_trans,
                                 obj_start, obj_end, obj_trans, wsl);
  k_final<<<1, 32, 0, stream>>>(wsl, (float*)d_out);
}
